// Joint_Region_Alignment_1726576857080
// MI455X (gfx1250) — matrix operand order repaired and checked by reading, not compiled, not run on hardware
//
#include <hip/hip_runtime.h>
#include <hip/hip_bf16.h>
#include <stdint.h>

// ---------------------------------------------------------------------------
// Joint Region Alignment — MI455X (gfx1250).
// All GEMM operands stored f16; v_wmma_f32_16x16x32_f16 with f32 accumulate.
// Double-buffered LDS staging uses GLOBAL_LOAD_ASYNC_TO_LDS_B128 when the
// toolchain exposes it (ASYNCcnt-tracked), else load+store fallback.
// Softmax / BN / LayerNorm math in fp32. Final output fp32.
// ---------------------------------------------------------------------------

typedef __attribute__((ext_vector_type(16))) _Float16 v16h;
typedef __attribute__((ext_vector_type(8)))  _Float16 h8;
typedef __attribute__((ext_vector_type(8)))  float    v8f;
typedef int v4i_ __attribute__((vector_size(16)));   // matches builtin param type

#define B_     8
#define E_     512
#define NI_    1024
#define NP_    1024
#define HEADS_ 4
#define HDIM_  128
#define EPS_   1e-5f
#define RS512  0.04419417382415922f   // 512^-0.5

#if defined(__has_builtin)
#  if __has_builtin(__builtin_amdgcn_global_load_async_to_lds_b128)
#    define HAS_ASYNC_LDS 1
#  endif
#endif
#ifndef HAS_ASYNC_LDS
#  define HAS_ASYNC_LDS 0
#endif

__device__ __forceinline__ void lds_move_b128(const _Float16* g, _Float16* l)
{
#if HAS_ASYNC_LDS
    __builtin_amdgcn_global_load_async_to_lds_b128((v4i_*)g, (v4i_*)l, 0, 0);
#else
    *(h8*)l = *(const h8*)g;
#endif
}

__device__ __forceinline__ void wait_async_lds()
{
#if HAS_ASYNC_LDS
#  if defined(__has_builtin) && __has_builtin(__builtin_amdgcn_s_wait_asynccnt)
    __builtin_amdgcn_s_wait_asynccnt(0);
#  else
    asm volatile("s_wait_asynccnt 0" ::: "memory");
#  endif
#endif
}

// ---------------------------------------------------------------------------
// Generic tiled WMMA GEMM:  C[M,N] = epilogue(A * B), A/B f16 in memory.
//   TA=0: A stored [M,K].  TA=1: A stored [K,M].
//   TB=0: B stored [K,N].  TB=1: B stored [N,K].
//   EPI: 0 none | 1 *alpha | 2 +p1[m] | 3 relu(acc*p1[m]+p2[m])
//   OUTH: 1 -> C is _Float16, 0 -> C is float.
// Tile BM=BN=128, BK=32; 256 threads = 8 waves; wave owns 2x4 16x16 subtiles.
// M,N multiples of 128; K multiple of 32; lda/ldb multiples of 8.
// ---------------------------------------------------------------------------
template <int TA, int TB, int EPI, int OUTH>
__launch_bounds__(256)
__global__ void gemm_wmma(const _Float16* __restrict__ A, long long lda, long long sA,
                          const _Float16* __restrict__ Bm, long long ldb, long long sB,
                          void* __restrict__ C, long long ldc, long long sC,
                          int M, int N, int K, float alpha,
                          const float* __restrict__ p1, const float* __restrict__ p2)
{
    constexpr int BM = 128, BN = 128, BK = 32, LD = BK + 8; // 40 halves = 80B rows
    constexpr bool anyAsync = HAS_ASYNC_LDS && ((!TA) || TB);
    __shared__ _Float16 As[2][BM * LD];
    __shared__ _Float16 Bs[2][BN * LD];

    const int tid    = threadIdx.x;
    const int lane   = tid & 31;
    const int wave   = tid >> 5;
    const int wm     = wave & 3;   // 4 wave rows -> 32 M-rows each
    const int wn     = wave >> 2;  // 2 wave cols -> 64 N-cols each
    const int lane15 = lane & 15;
    const int laneHi = lane >> 4;

    const long long bz = blockIdx.z;
    const int m0 = blockIdx.y * BM;
    const int n0 = blockIdx.x * BN;
    const _Float16* Ab = A  + bz * sA;
    const _Float16* Bb = Bm + bz * sB;

    auto stageA = [&](int buf, int k0) {
        if constexpr (!TA) {          // [M,K]: b128 along k -> async straight to LDS
            #pragma unroll
            for (int it = 0; it < (BM * BK / 8) / 256; ++it) {
                int v  = tid + it * 256;
                int r  = v >> 2;                 // BK/8 = 4 chunks per row
                int kj = (v & 3) << 3;
                lds_move_b128(Ab + (long long)(m0 + r) * lda + (k0 + kj),
                              &As[buf][r * LD + kj]);
            }
        } else {                      // [K,M]: b128 along m, scatter-transpose
            #pragma unroll
            for (int it = 0; it < (BK * BM / 8) / 256; ++it) {
                int v  = tid + it * 256;
                int k  = v >> 4;                 // BM/8 = 16 chunks per k-row
                int mj = (v & 15) << 3;
                h8 q = *(const h8*)(Ab + (long long)(k0 + k) * lda + (m0 + mj));
                #pragma unroll
                for (int e = 0; e < 8; ++e) As[buf][(mj + e) * LD + k] = q[e];
            }
        }
    };
    auto stageB = [&](int buf, int k0) {
        if constexpr (!TB) {          // [K,N]: b128 along n, scatter -> Bs[n][k]
            #pragma unroll
            for (int it = 0; it < (BK * BN / 8) / 256; ++it) {
                int v  = tid + it * 256;
                int k  = v >> 4;                 // BN/8 = 16 chunks per k-row
                int nj = (v & 15) << 3;
                h8 q = *(const h8*)(Bb + (long long)(k0 + k) * ldb + (n0 + nj));
                #pragma unroll
                for (int e = 0; e < 8; ++e) Bs[buf][(nj + e) * LD + k] = q[e];
            }
        } else {                      // [N,K]: b128 along k -> async straight to LDS
            #pragma unroll
            for (int it = 0; it < (BN * BK / 8) / 256; ++it) {
                int v  = tid + it * 256;
                int n  = v >> 2;                 // BK/8 = 4 chunks per row
                int kj = (v & 3) << 3;
                lds_move_b128(Bb + (long long)(n0 + n) * ldb + (k0 + kj),
                              &Bs[buf][n * LD + kj]);
            }
        }
    };

    v8f acc[2][4];
    #pragma unroll
    for (int i = 0; i < 2; ++i)
        #pragma unroll
        for (int j = 0; j < 4; ++j) acc[i][j] = {};

    stageA(0, 0);
    stageB(0, 0);
    if constexpr (anyAsync) wait_async_lds();
    __syncthreads();

    const int nk = K / BK;
    for (int kb = 0; kb < nk; ++kb) {
        const int cur = kb & 1;
        if (kb + 1 < nk) { stageA(cur ^ 1, (kb + 1) * BK); stageB(cur ^ 1, (kb + 1) * BK); }
        if (kb + 2 < nk) {
            const int kp = (kb + 2) * BK;
            if constexpr (!TA) __builtin_prefetch(&Ab[(long long)(m0 + (tid & 127)) * lda + kp], 0, 1);
            else               __builtin_prefetch(&Ab[(long long)(kp + (tid & 31)) * lda + m0 + ((tid >> 5) << 4)], 0, 1);
            if constexpr (!TB) __builtin_prefetch(&Bb[(long long)(kp + (tid & 31)) * ldb + n0 + ((tid >> 5) << 4)], 0, 1);
            else               __builtin_prefetch(&Bb[(long long)(n0 + (tid & 127)) * ldb + kp], 0, 1);
        }

        // A frag (16x32): lanes 0-15 k0-15, lanes 16-31 k16-31
        v16h af[2];
        #pragma unroll
        for (int ms = 0; ms < 2; ++ms) {
            const _Float16* row = &As[cur][(wm * 32 + ms * 16 + lane15) * LD];
            const int kb0 = laneHi ? 16 : 0;
            #pragma unroll
            for (int i = 0; i < 8; ++i) {
                af[ms][i]     = row[kb0 + i];
                af[ms][8 + i] = row[kb0 + 8 + i];
            }
        }
        // B frag (32x16): lanes 0-15 n=lane k0-15, lanes 16-31 n=lane-16 k16-31
        v16h bf[4];
        #pragma unroll
        for (int ns = 0; ns < 4; ++ns) {
            const _Float16* col = &Bs[cur][(wn * 64 + ns * 16 + lane15) * LD + (laneHi ? 16 : 0)];
            #pragma unroll
            for (int i = 0; i < 16; ++i) bf[ns][i] = col[i];
        }
        #pragma unroll
        for (int ms = 0; ms < 2; ++ms)
            #pragma unroll
            for (int ns = 0; ns < 4; ++ns)
                acc[ms][ns] = __builtin_amdgcn_wmma_f32_16x16x32_f16(
                    false, af[ms], false, bf[ns], (short)0, acc[ms][ns], false, false);

        if constexpr (anyAsync) wait_async_lds();   // staged next buffer complete
        __syncthreads();
    }

    // C layout: lanes 0-15 N=lane M=r; lanes 16-31 N=lane-16 M=r+8
    #pragma unroll
    for (int ms = 0; ms < 2; ++ms) {
        const int mb = m0 + wm * 32 + ms * 16 + laneHi * 8;
        #pragma unroll
        for (int ns = 0; ns < 4; ++ns) {
            const int n = n0 + wn * 64 + ns * 16 + lane15;
            #pragma unroll
            for (int r = 0; r < 8; ++r) {
                const int m = mb + r;
                float v = acc[ms][ns][r];
                if constexpr (EPI == 1) v *= alpha;
                else if constexpr (EPI == 2) v += p1[m];
                else if constexpr (EPI == 3) { v = v * p1[m] + p2[m]; v = v > 0.f ? v : 0.f; }
                if constexpr (OUTH) ((_Float16*)C)[bz * sC + (long long)m * ldc + n] = (_Float16)v;
                else                ((float*)C)[bz * sC + (long long)m * ldc + n] = v;
            }
        }
    }
}

// --------------------------- fp32 helper kernels ---------------------------

__launch_bounds__(256)
__global__ void cvt_f32_f16(const float* __restrict__ src, _Float16* __restrict__ dst,
                            long long n)
{
    long long i = ((long long)blockIdx.x * 256 + threadIdx.x) * 4;
    if (i + 3 < n) {
        const float4 q = *(const float4*)(src + i);
        dst[i]     = (_Float16)q.x; dst[i + 1] = (_Float16)q.y;
        dst[i + 2] = (_Float16)q.z; dst[i + 3] = (_Float16)q.w;
    }
}

// fold conv bias + eval BatchNorm into scale/shift per channel
__launch_bounds__(256)
__global__ void bn_prep(const float* b1, const float* g1, const float* bt1,
                        const float* m1, const float* v1,
                        const float* b2, const float* g2, const float* bt2,
                        const float* m2, const float* v2,
                        float* s1, float* sh1, float* s2, float* sh2)
{
    const int t = blockIdx.x * 256 + threadIdx.x;
    if (t < 2 * E_) {
        float sc = g1[t] * rsqrtf(v1[t] + EPS_);
        s1[t] = sc; sh1[t] = (b1[t] - m1[t]) * sc + bt1[t];
    }
    if (t < E_) {
        float sc = g2[t] * rsqrtf(v2[t] + EPS_);
        s2[t] = sc; sh2[t] = (b2[t] - m2[t]) * sc + bt2[t];
    }
}

// softmax over last axis; one block per row; f16 output
template <typename Tin>
__launch_bounds__(256)
__global__ void softmax_row(const Tin* __restrict__ src, _Float16* __restrict__ dst, int cols)
{
    const long long row = blockIdx.x;
    const Tin*      s = src + row * (long long)cols;
    _Float16*       d = dst + row * (long long)cols;
    __shared__ float red[8];
    const int tid = threadIdx.x, lane = tid & 31, wv = tid >> 5;

    float m = -3.4e38f;
    for (int c = tid; c < cols; c += 256) m = fmaxf(m, (float)s[c]);
    #pragma unroll
    for (int o = 16; o > 0; o >>= 1) m = fmaxf(m, __shfl_xor(m, o, 32));
    if (lane == 0) red[wv] = m;
    __syncthreads();
    float bm = red[0];
    #pragma unroll
    for (int i = 1; i < 8; ++i) bm = fmaxf(bm, red[i]);
    __syncthreads();

    float sum = 0.f;
    for (int c = tid; c < cols; c += 256) {
        float e = __expf((float)s[c] - bm);
        d[c] = (_Float16)e; sum += e;
    }
    #pragma unroll
    for (int o = 16; o > 0; o >>= 1) sum += __shfl_xor(sum, o, 32);
    if (lane == 0) red[wv] = sum;
    __syncthreads();
    float tot = 0.f;
    #pragma unroll
    for (int i = 0; i < 8; ++i) tot += red[i];
    const float inv = 1.f / tot;
    for (int c = tid; c < cols; c += 256) d[c] = (_Float16)((float)d[c] * inv);
}

// softmax over axis 1 of [B, P, I] (f16 in/out); block = 64 cols x 4 p-lanes
__launch_bounds__(256)
__global__ void softmax_col(const _Float16* __restrict__ src, _Float16* __restrict__ dst,
                            int P, int I)
{
    const int b  = blockIdx.y;
    const int c  = threadIdx.x & 63;
    const int i  = blockIdx.x * 64 + c;
    const int pr = threadIdx.x >> 6;
    const _Float16* s = src + (long long)b * P * I + i;
    _Float16*       d = dst + (long long)b * P * I + i;
    __shared__ float red[4][64];

    float m = -3.4e38f;
    for (int p = pr; p < P; p += 4) m = fmaxf(m, (float)s[(long long)p * I]);
    red[pr][c] = m; __syncthreads();
    m = fmaxf(fmaxf(red[0][c], red[1][c]), fmaxf(red[2][c], red[3][c]));
    __syncthreads();

    float sum = 0.f;
    for (int p = pr; p < P; p += 4) {
        float e = __expf((float)s[(long long)p * I] - m);
        d[(long long)p * I] = (_Float16)e; sum += e;
    }
    red[pr][c] = sum; __syncthreads();
    sum = red[0][c] + red[1][c] + red[2][c] + red[3][c];
    const float inv = 1.f / sum;
    for (int p = pr; p < P; p += 4)
        d[(long long)p * I] = (_Float16)((float)d[(long long)p * I] * inv);
}

// y = attn(f32) + x(f16) ; LayerNorm(y)*g+b -> optional f32 dst and/or f16 dst
__launch_bounds__(256)
__global__ void residual_ln(const float* __restrict__ attn, long long attn_b,
                            const _Float16* __restrict__ x, long long xs, long long xe,
                            long long x_b,
                            const float* __restrict__ g, const float* __restrict__ be,
                            float* __restrict__ dstF, _Float16* __restrict__ dstH,
                            long long dst_b, int s_off)
{
    const int b = blockIdx.y, s = blockIdx.x, tid = threadIdx.x;
    const int lane = tid & 31, wv = tid >> 5;
    const float*    ap = attn + (long long)b * attn_b + (long long)s * E_;
    const _Float16* xp = x    + (long long)b * x_b    + (long long)s * xs;

    float y0 = ap[tid]       + (float)xp[(long long)tid * xe];
    float y1 = ap[tid + 256] + (float)xp[(long long)(tid + 256) * xe];
    float ls = y0 + y1, lq = y0 * y0 + y1 * y1;
    #pragma unroll
    for (int o = 16; o > 0; o >>= 1) { ls += __shfl_xor(ls, o, 32); lq += __shfl_xor(lq, o, 32); }
    __shared__ float rs[8], rq[8];
    if (lane == 0) { rs[wv] = ls; rq[wv] = lq; }
    __syncthreads();
    float S = 0.f, Q = 0.f;
    #pragma unroll
    for (int i = 0; i < 8; ++i) { S += rs[i]; Q += rq[i]; }
    const float mu  = S * (1.f / E_);
    const float var = Q * (1.f / E_) - mu * mu;
    const float inv = rsqrtf(var + EPS_);
    const float o0 = (y0 - mu) * inv * g[tid]       + be[tid];
    const float o1 = (y1 - mu) * inv * g[tid + 256] + be[tid + 256];
    const long long dof = (long long)b * dst_b + (long long)(s + s_off) * E_;
    if (dstF) { dstF[dof + tid] = o0; dstF[dof + tid + 256] = o1; }
    if (dstH) { dstH[dof + tid] = (_Float16)o0; dstH[dof + tid + 256] = (_Float16)o1; }
}

// ----------------------------- host orchestration --------------------------

static void run_mha(hipStream_t st, const _Float16* hWq, const float* bq,
                    const _Float16* hWk, const float* bk,
                    const _Float16* hWv, const float* bv,
                    const float* lng, const float* lnb,
                    const _Float16* x, bool xRowMajor, int S,
                    _Float16* hQt, _Float16* hKt, _Float16* hVt,
                    float* scores, _Float16* probs, float* attn,
                    float* dstF, _Float16* dstH, int s_off, long long dst_b)
{
    const long long xstride = (long long)E_ * S;
    const long long qs = (long long)E_ * S;
    const dim3 gp(S / 128, E_ / 128, B_);
    if (!xRowMajor) { // x = [B, E, S]
        gemm_wmma<0,0,2,1><<<gp,256,0,st>>>(hWq,E_,0, x,S,xstride, hQt,S,qs, E_,S,E_, 1.f, bq, nullptr);
        gemm_wmma<0,0,2,1><<<gp,256,0,st>>>(hWk,E_,0, x,S,xstride, hKt,S,qs, E_,S,E_, 1.f, bk, nullptr);
        gemm_wmma<0,0,2,1><<<gp,256,0,st>>>(hWv,E_,0, x,S,xstride, hVt,S,qs, E_,S,E_, 1.f, bv, nullptr);
    } else {          // x = [B, S, E] row-major -> TB
        gemm_wmma<0,1,2,1><<<gp,256,0,st>>>(hWq,E_,0, x,E_,xstride, hQt,S,qs, E_,S,E_, 1.f, bq, nullptr);
        gemm_wmma<0,1,2,1><<<gp,256,0,st>>>(hWk,E_,0, x,E_,xstride, hKt,S,qs, E_,S,E_, 1.f, bk, nullptr);
        gemm_wmma<0,1,2,1><<<gp,256,0,st>>>(hWv,E_,0, x,E_,xstride, hVt,S,qs, E_,S,E_, 1.f, bv, nullptr);
    }
    const long long hs = (long long)HDIM_ * S;   // head slice stride in Qt/Kt/Vt
    for (int b = 0; b < B_; ++b) {
        const _Float16* qb  = hQt + (long long)b * qs;
        const _Float16* kb2 = hKt + (long long)b * qs;
        const _Float16* vb  = hVt + (long long)b * qs;
        float* ob = attn + (long long)b * S * E_;
        // scores[h][s,t] = scale * sum_d Qt[h*128+d, s] * Kt[h*128+d, t]
        gemm_wmma<1,0,1,0><<<dim3(S/128,S/128,HEADS_),256,0,st>>>(
            qb,S,hs, kb2,S,hs, scores,S,(long long)S*S, S,S,HDIM_, RS512, nullptr, nullptr);
        softmax_row<float><<<HEADS_*S,256,0,st>>>(scores, probs, S);
        // out[s, h*128+d] = sum_t P[h][s,t] * Vt[h*128+d, t]
        gemm_wmma<0,1,0,0><<<dim3(1,S/128,HEADS_),256,0,st>>>(
            probs,S,(long long)S*S, vb,S,hs, ob,E_,HDIM_, S,HDIM_,S, 1.f, nullptr, nullptr);
    }
    const long long xsr = xRowMajor ? E_ : 1;
    const long long xer = xRowMajor ? 1 : S;
    residual_ln<<<dim3(S,B_),256,0,st>>>(attn,(long long)S*E_, x,xsr,xer,xstride,
                                         lng,lnb, dstF,dstH, dst_b, s_off);
}

extern "C" void kernel_launch(void* const* d_in, const int* in_sizes, int n_in,
                              void* d_out, int out_size, void* d_ws, size_t ws_size,
                              hipStream_t stream)
{
    (void)in_sizes; (void)n_in; (void)out_size; (void)ws_size;
    const float* F_i = (const float*)d_in[0];
    const float* F_p = (const float*)d_in[1];
    const float* W1  = (const float*)d_in[2];
    const float* cb1 = (const float*)d_in[3];
    const float* g1  = (const float*)d_in[4];
    const float* bt1 = (const float*)d_in[5];
    const float* m1  = (const float*)d_in[6];
    const float* v1  = (const float*)d_in[7];
    const float* W2  = (const float*)d_in[8];
    const float* cb2 = (const float*)d_in[9];
    const float* g2  = (const float*)d_in[10];
    const float* bt2 = (const float*)d_in[11];
    const float* m2  = (const float*)d_in[12];
    const float* v2  = (const float*)d_in[13];
    const float* const* ai = (const float* const*)&d_in[14]; // Wq,bq,Wk,bk,Wv,bv,lng,lnb
    const float* const* ap = (const float* const*)&d_in[22];
    const float* const* aj = (const float* const*)&d_in[30];
    float* out = (float*)d_out;                              // [B, 2048, 512]

    // ---- workspace bump allocator (256B aligned) ----
    char* base = (char*)d_ws;
    size_t off = 0;
    auto alloc = [&](size_t bytes) -> char* {
        char* p = base + off; off += (bytes + 255) & ~(size_t)255; return p;
    };
    const size_t Mi = 1u << 20;
    _Float16* hFi   = (_Float16*)alloc(4 * Mi * 2);          // [B,E,1024]
    _Float16* hFp   = (_Float16*)alloc(4 * Mi * 2);
    _Float16* hW1   = (_Float16*)alloc((size_t)2 * E_ * E_ * 2);
    _Float16* hW2   = (_Float16*)alloc((size_t)2 * E_ * E_ * 2);
    _Float16* hWa[9];
    for (int i = 0; i < 9; ++i) hWa[i] = (_Float16*)alloc((size_t)E_ * E_ * 2);
    _Float16* hH1   = (_Float16*)alloc(8 * Mi * 2);          // [B,2E,1024]
    _Float16* hI    = (_Float16*)alloc(4 * Mi * 2);          // [B,E,NI]
    _Float16* hP    = (_Float16*)alloc(4 * Mi * 2);
    _Float16* hphi  = (_Float16*)alloc(8 * Mi * 2);          // [B,NP,NI]
    _Float16* hphiP = (_Float16*)alloc(8 * Mi * 2);
    _Float16* hphiI = (_Float16*)alloc(8 * Mi * 2);
    _Float16* hIenh = (_Float16*)alloc(4 * Mi * 2);          // [B,E,NI]
    _Float16* hPenh = (_Float16*)alloc(4 * Mi * 2);
    _Float16* hJnt  = (_Float16*)alloc(8 * Mi * 2);          // [B,2048,E]
    _Float16* hQt   = (_Float16*)alloc(8 * Mi * 2);          // [B,E,S<=2048]
    _Float16* hKt   = (_Float16*)alloc(8 * Mi * 2);
    _Float16* hVt   = (_Float16*)alloc(8 * Mi * 2);
    _Float16* probs = (_Float16*)alloc(16 * Mi * 2);         // [4, S, S] per batch
    float*    scores= (float*)   alloc(16 * Mi * 4);         // [4, S, S] per batch
    float*    attn  = (float*)   alloc(8 * Mi * 4);          // [B, S, E]
    float*    s1    = (float*)   alloc(4096 * 4);
    float* sh1 = s1 + 1024; float* s2 = sh1 + 1024; float* sh2 = s2 + 512;

    // ---- one-time f32 -> f16 conversions ----
    auto cvt = [&](const float* s, _Float16* d, long long n) {
        cvt_f32_f16<<<(unsigned)((n / 4 + 255) / 256), 256, 0, stream>>>(s, d, n);
    };
    cvt(F_i, hFi, 4 * (long long)Mi);
    cvt(F_p, hFp, 4 * (long long)Mi);
    cvt(W1,  hW1, (long long)2 * E_ * E_);
    cvt(W2,  hW2, (long long)2 * E_ * E_);
    const float* const* sets[3] = { ai, ap, aj };
    for (int s = 0; s < 3; ++s)
        for (int w = 0; w < 3; ++w)
            cvt(sets[s][2 * w], hWa[s * 3 + w], (long long)E_ * E_);

    bn_prep<<<4,256,0,stream>>>(cb1,g1,bt1,m1,v1, cb2,g2,bt2,m2,v2, s1,sh1,s2,sh2);

    // ---- _to_common(F_i) -> hI ; _to_common(F_p) -> hP ----
    gemm_wmma<0,0,3,1><<<dim3(8,8,B_),256,0,stream>>>(
        hW1,E_,0, hFi,NI_,(long long)E_*NI_, hH1,NI_,(long long)2*E_*NI_,
        2*E_,NI_,E_, 1.f, s1, sh1);
    gemm_wmma<0,0,3,1><<<dim3(8,4,B_),256,0,stream>>>(
        hW2,2*E_,0, hH1,NI_,(long long)2*E_*NI_, hI,NI_,(long long)E_*NI_,
        E_,NI_,2*E_, 1.f, s2, sh2);
    gemm_wmma<0,0,3,1><<<dim3(8,8,B_),256,0,stream>>>(
        hW1,E_,0, hFp,NP_,(long long)E_*NP_, hH1,NP_,(long long)2*E_*NP_,
        2*E_,NP_,E_, 1.f, s1, sh1);
    gemm_wmma<0,0,3,1><<<dim3(8,4,B_),256,0,stream>>>(
        hW2,2*E_,0, hH1,NP_,(long long)2*E_*NP_, hP,NP_,(long long)E_*NP_,
        E_,NP_,2*E_, 1.f, s2, sh2);

    // ---- phi[p,i] = scale * sum_c P[c,p] I[c,i] ----
    gemm_wmma<1,0,1,1><<<dim3(8,8,B_),256,0,stream>>>(
        hP,NP_,(long long)E_*NP_, hI,NI_,(long long)E_*NI_,
        hphi,NI_,(long long)NP_*NI_, NP_,NI_,E_, RS512, nullptr, nullptr);

    softmax_col<<<dim3(NI_/64,B_),256,0,stream>>>(hphi, hphiP, NP_, NI_); // axis=1
    softmax_row<_Float16><<<B_*NP_,256,0,stream>>>(hphi, hphiI, NI_);     // axis=2

    // ---- I_enh = P @ phiP ; P_enh = I @ phiI^T ----
    gemm_wmma<0,0,0,1><<<dim3(8,4,B_),256,0,stream>>>(
        hP,NP_,(long long)E_*NP_, hphiP,NI_,(long long)NP_*NI_,
        hIenh,NI_,(long long)E_*NI_, E_,NI_,NP_, 1.f, nullptr, nullptr);
    gemm_wmma<0,1,0,1><<<dim3(8,4,B_),256,0,stream>>>(
        hI,NI_,(long long)E_*NI_, hphiI,NI_,(long long)NP_*NI_,
        hPenh,NP_,(long long)E_*NP_, E_,NP_,NI_, 1.f, nullptr, nullptr);

    // ---- MHA(P_enh) -> joint rows [0,1024) ; MHA(I_enh) -> rows [1024,2048) ----
    run_mha(stream, hWa[3],ap[1], hWa[4],ap[3], hWa[5],ap[5], ap[6],ap[7],
            hPenh, false, NP_, hQt,hKt,hVt, scores,probs,attn,
            nullptr, hJnt, 0, 2048LL * E_);
    run_mha(stream, hWa[0],ai[1], hWa[1],ai[3], hWa[2],ai[5], ai[6],ai[7],
            hIenh, false, NI_, hQt,hKt,hVt, scores,probs,attn,
            nullptr, hJnt, 1024, 2048LL * E_);
    // ---- joint MHA -> final fp32 output ----
    run_mha(stream, hWa[6],aj[1], hWa[7],aj[3], hWa[8],aj[5], aj[6],aj[7],
            hJnt, true, 2048, hQt,hKt,hVt, scores,probs,attn,
            out, nullptr, 0, 2048LL * E_);
}
